// GRU_57578331570464
// MI455X (gfx1250) — matrix operand order repaired and checked by reading, not compiled, not run on hardware
//
#include <hip/hip_runtime.h>
#include <hip/hip_bf16.h>

// ---------- types ----------
typedef __bf16 bf16_t;
typedef __attribute__((ext_vector_type(16))) __bf16 bf16x16;
typedef __attribute__((ext_vector_type(8)))  __bf16 bf16x8;
typedef __attribute__((ext_vector_type(8)))  float  f32x8;

#define Bsz 512
#define Tsz 365
#define INs 10
#define Hs  256
#define OUTs 20
#define G3  768            // 3*H
#define KT  8              // 256/32 k-steps
#define NT3 48             // 768/16 n-tiles

// ---------- bf16 helpers (bit-exact, no cvt dependence) ----------
__device__ inline bf16_t f2bf(float x) {
    union { float f; unsigned u; } v; v.f = x;
    unsigned r = v.u + 0x7FFFu + ((v.u >> 16) & 1u);   // round-nearest-even
    unsigned short h = (unsigned short)(r >> 16);
    return __builtin_bit_cast(bf16_t, h);
}
__device__ inline float bf2f(bf16_t x) {
    unsigned short h = __builtin_bit_cast(unsigned short, x);
    union { unsigned u; float f; } v; v.u = ((unsigned)h) << 16;
    return v.f;
}
__device__ inline float sigmoidf_(float x) { return 1.0f / (1.0f + __expf(-x)); }

// ---------- WMMA ----------
__device__ inline f32x8 wmma_bf16(bf16x16 a, bf16x16 b, f32x8 c) {
    return __builtin_amdgcn_wmma_f32_16x16x32_bf16(false, a, false, b, (short)0, c, false, false);
}

// A fragment: 16x32 bf16 tile of row-major A[., lda], per ISA 16-bit A layout.
// lane 0-15  : row = mbase+lane,    halves 0-7 -> K=kb+0..7,  8-15 -> K=kb+16..23
// lane 16-31 : row = mbase+lane-16, halves 0-7 -> K=kb+8..15, 8-15 -> K=kb+24..31
__device__ inline bf16x16 load_a_frag(const bf16_t* __restrict__ A, int lda,
                                      int mbase, int kb, int lane) {
    int row = mbase + (lane & 15);
    int k0  = kb + ((lane & 16) ? 8 : 0);
    const bf16x8* p0 = (const bf16x8*)(A + row * lda + k0);
    const bf16x8* p1 = (const bf16x8*)(A + row * lda + k0 + 16);
    bf16x8 lo = *p0, hi = *p1;
    bf16x16 f;
#pragma unroll
    for (int i = 0; i < 8; ++i) { f[i] = lo[i]; f[i + 8] = hi[i]; }
    return f;
}

// B fragment from prepacked weights: one contiguous 32B load per lane.
__device__ inline bf16x16 load_b_frag(const bf16_t* __restrict__ Wp, int ntile, int kt, int lane) {
    return *(const bf16x16*)(Wp + ((((ntile * KT) + kt) * 32 + lane) << 4));
}

// 3-gate GEMM with software-pipelined A loads: while WMMAs for k-step kt run on
// the matrix pipe, the two global_load_b128 for kt+1 are already in flight.
__device__ inline void gate_gemm3(const bf16_t* __restrict__ A, const bf16_t* __restrict__ Wp,
                                  int mbase, int jt, int lane,
                                  f32x8& ar, f32x8& az, f32x8& an) {
    f32x8 zr = {0.f,0.f,0.f,0.f,0.f,0.f,0.f,0.f};
    ar = zr; az = zr; an = zr;
    bf16x16 a_cur = load_a_frag(A, Hs, mbase, 0, lane);
#pragma unroll
    for (int kt = 0; kt < KT - 1; ++kt) {
        bf16x16 a_nxt = load_a_frag(A, Hs, mbase, (kt + 1) * 32, lane);
        bf16x16 br = load_b_frag(Wp,      jt, kt, lane);
        bf16x16 bz = load_b_frag(Wp, 16 + jt, kt, lane);
        bf16x16 bn = load_b_frag(Wp, 32 + jt, kt, lane);
        ar = wmma_bf16(a_cur, br, ar);
        az = wmma_bf16(a_cur, bz, az);
        an = wmma_bf16(a_cur, bn, an);
        a_cur = a_nxt;
    }
    {
        bf16x16 br = load_b_frag(Wp,      jt, KT - 1, lane);
        bf16x16 bz = load_b_frag(Wp, 16 + jt, KT - 1, lane);
        bf16x16 bn = load_b_frag(Wp, 32 + jt, KT - 1, lane);
        ar = wmma_bf16(a_cur, br, ar);
        az = wmma_bf16(a_cur, bz, az);
        an = wmma_bf16(a_cur, bn, an);
    }
}

// ---------- grid-wide barrier (persistent kernels, monotonic counter) ----------
__device__ inline void grid_barrier(unsigned* cnt, unsigned step, unsigned nb) {
    __syncthreads();
    if (threadIdx.x == 0) {
        __threadfence();
        atomicAdd(cnt, 1u);
        unsigned target = nb * (step + 1u);
        while (__atomic_load_n(cnt, __ATOMIC_RELAXED) < target)
            __builtin_amdgcn_s_sleep(2);
    }
    __syncthreads();
    __threadfence();
}

// ---------- weight prepack: f32 [768,256] -> bf16 WMMA B-fragment order ----------
__global__ void __launch_bounds__(256) pack_w(const float* __restrict__ W, bf16_t* __restrict__ Wp) {
    int idx = blockIdx.x * 256 + threadIdx.x;          // 48*8*32*16 = 196608
    if (idx >= NT3 * KT * 32 * 16) return;
    int q     =  idx        & 15;
    int lane  = (idx >> 4)  & 31;
    int kt    = (idx >> 9)  & 7;
    int ntile =  idx >> 12;
    int n = ntile * 16 + (lane & 15);
    int k = kt * 32 + ((lane & 16) ? 8 : 0) + (q & 7) + ((q & 8) ? 16 : 0);
    Wp[idx] = f2bf(W[n * Hs + k]);
}

// ---------- layer 0 persistent recurrence (64 blocks x 256 thr = 512 waves) ----------
__global__ void __launch_bounds__(256) gru_layer0(
        const float* __restrict__ x, const float* __restrict__ Wih0,
        const float* __restrict__ bih0, const float* __restrict__ bhh0,
        const bf16_t* __restrict__ WpHH0,
        bf16_t* __restrict__ hb0, bf16_t* __restrict__ hb1,
        bf16_t* __restrict__ ys0, unsigned* cnt) {
    int lane = threadIdx.x & 31;
    int wave = (blockIdx.x << 3) | (threadIdx.x >> 5);
    int mt = wave >> 4, jt = wave & 15;
    int mbase = mt * 16;
    int col = jt * 16 + (lane & 15);
    int rowoff = (lane & 16) ? 8 : 0;
    for (int t = 0; t < Tsz; ++t) {
        const bf16_t* hprev = (t & 1) ? hb1 : hb0;
        bf16_t*       hnext = (t & 1) ? hb0 : hb1;
        f32x8 ar, az, an;
        gate_gemm3(hprev, WpHH0, mbase, jt, lane, ar, az, an);
#pragma unroll
        for (int e = 0; e < 8; ++e) {
            int b = mbase + rowoff + e;
            const float* xp = x + (b * Tsz + t) * INs;
            float gr = bih0[col], gz = bih0[Hs + col], gn = bih0[2 * Hs + col];
#pragma unroll
            for (int k = 0; k < INs; ++k) {
                float xv = xp[k];
                gr += xv * Wih0[col * INs + k];
                gz += xv * Wih0[(Hs + col) * INs + k];
                gn += xv * Wih0[(2 * Hs + col) * INs + k];
            }
            float r = sigmoidf_(gr + ar[e] + bhh0[col]);
            float z = sigmoidf_(gz + az[e] + bhh0[Hs + col]);
            float n = tanhf(gn + r * (an[e] + bhh0[2 * Hs + col]));
            float hv = bf2f(hprev[b * Hs + col]);
            bf16_t hb = f2bf((1.f - z) * n + z * hv);
            hnext[b * Hs + col] = hb;
            ys0[(t * Bsz + b) * Hs + col] = hb;
        }
        grid_barrier(cnt, (unsigned)t, gridDim.x);
    }
}

// ---------- fully-parallel GI1 = ys0 @ W_ih1^T + b_ih1 ----------
__global__ void __launch_bounds__(256) gi1_gemm(
        const bf16_t* __restrict__ ys0, const bf16_t* __restrict__ WpIH1,
        const float* __restrict__ bih1, bf16_t* __restrict__ gi1) {
    int lane = threadIdx.x & 31;
    int wave = (blockIdx.x << 3) | (threadIdx.x >> 5);
    int mt = wave >> 4, jt = wave & 15;                // mt: 0..11679
    if (mt >= (Tsz * Bsz) / 16) return;
    f32x8 ar, az, an;
    gate_gemm3(ys0, WpIH1, mt * 16, jt, lane, ar, az, an);
    int col = jt * 16 + (lane & 15);
    int rowoff = (lane & 16) ? 8 : 0;
#pragma unroll
    for (int e = 0; e < 8; ++e) {
        int row = mt * 16 + rowoff + e;                // row = t*B + b
        gi1[row * G3 +           col] = f2bf(ar[e] + bih1[col]);
        gi1[row * G3 + Hs      + col] = f2bf(az[e] + bih1[Hs + col]);
        gi1[row * G3 + 2 * Hs  + col] = f2bf(an[e] + bih1[2 * Hs + col]);
    }
}

// ---------- layer 1 persistent recurrence ----------
__global__ void __launch_bounds__(256) gru_layer1(
        const bf16_t* __restrict__ gi1, const float* __restrict__ bhh1,
        const bf16_t* __restrict__ WpHH1,
        bf16_t* __restrict__ hb0, bf16_t* __restrict__ hb1, unsigned* cnt) {
    int lane = threadIdx.x & 31;
    int wave = (blockIdx.x << 3) | (threadIdx.x >> 5);
    int mt = wave >> 4, jt = wave & 15;
    int mbase = mt * 16;
    int col = jt * 16 + (lane & 15);
    int rowoff = (lane & 16) ? 8 : 0;
    for (int t = 0; t < Tsz; ++t) {
        const bf16_t* hprev = (t & 1) ? hb1 : hb0;
        bf16_t*       hnext = (t & 1) ? hb0 : hb1;
        f32x8 ar, az, an;
        gate_gemm3(hprev, WpHH1, mbase, jt, lane, ar, az, an);
#pragma unroll
        for (int e = 0; e < 8; ++e) {
            int b = mbase + rowoff + e;
            const bf16_t* gp = gi1 + (size_t)(t * Bsz + b) * G3;
            float r = sigmoidf_(bf2f(gp[col])           + ar[e] + bhh1[col]);
            float z = sigmoidf_(bf2f(gp[Hs + col])      + az[e] + bhh1[Hs + col]);
            float n = tanhf(bf2f(gp[2 * Hs + col]) + r * (an[e] + bhh1[2 * Hs + col]));
            float hv = bf2f(hprev[b * Hs + col]);
            hnext[b * Hs + col] = f2bf((1.f - z) * n + z * hv);
        }
        grid_barrier(cnt, (unsigned)t, gridDim.x);
    }
}

// ---------- classifier + softmax (one wave per batch row) ----------
__global__ void __launch_bounds__(32) classify(
        const bf16_t* __restrict__ hlast, const float* __restrict__ Wout,
        const float* __restrict__ bout, float* __restrict__ out) {
    int b = blockIdx.x;
    int l = threadIdx.x;
    float v = -3.0e38f;
    if (l < OUTs) {
        float s = bout[l];
        for (int k = 0; k < Hs; ++k) s += bf2f(hlast[b * Hs + k]) * Wout[l * Hs + k];
        v = s;
    }
    float m = v;
#pragma unroll
    for (int o = 16; o; o >>= 1) m = fmaxf(m, __shfl_xor(m, o, 32));
    float e = (l < OUTs) ? __expf(v - m) : 0.f;
    float sum = e;
#pragma unroll
    for (int o = 16; o; o >>= 1) sum += __shfl_xor(sum, o, 32);
    if (l < OUTs) out[b * OUTs + l] = e / sum;
}

// ---------- launch ----------
extern "C" void kernel_launch(void* const* d_in, const int* in_sizes, int n_in,
                              void* d_out, int out_size, void* d_ws, size_t ws_size,
                              hipStream_t stream) {
    const float* x    = (const float*)d_in[0];
    const float* Wih0 = (const float*)d_in[3];
    const float* Whh0 = (const float*)d_in[4];
    const float* bih0 = (const float*)d_in[5];
    const float* bhh0 = (const float*)d_in[6];
    const float* Wih1 = (const float*)d_in[7];
    const float* Whh1 = (const float*)d_in[8];
    const float* bih1 = (const float*)d_in[9];
    const float* bhh1 = (const float*)d_in[10];
    const float* Wout = (const float*)d_in[11];
    const float* bout = (const float*)d_in[12];

    char* ws = (char*)d_ws;
    const size_t HBUF = (size_t)Bsz * Hs * sizeof(bf16_t);        // 256 KB
    unsigned* cnt0 = (unsigned*)(ws + 0);
    unsigned* cnt1 = (unsigned*)(ws + 256);
    bf16_t* h0a = (bf16_t*)(ws + 1024);
    bf16_t* h0b = (bf16_t*)(ws + 1024 + 1 * HBUF);
    bf16_t* h1a = (bf16_t*)(ws + 1024 + 2 * HBUF);
    bf16_t* h1b = (bf16_t*)(ws + 1024 + 3 * HBUF);
    size_t off = 1024 + 4 * HBUF;
    const size_t WPK = (size_t)G3 * Hs * sizeof(bf16_t);          // 384 KB
    bf16_t* WpHH0 = (bf16_t*)(ws + off);
    bf16_t* WpIH1 = (bf16_t*)(ws + off + WPK);
    bf16_t* WpHH1 = (bf16_t*)(ws + off + 2 * WPK);
    off += 3 * WPK;
    bf16_t* ys0 = (bf16_t*)(ws + off);                            // ~96 MB
    off += (size_t)Tsz * Bsz * Hs * sizeof(bf16_t);
    bf16_t* gi1 = (bf16_t*)(ws + off);                            // ~287 MB

    // zero counters + initial hidden states (h0 = 0)
    hipMemsetAsync(ws, 0, 1024 + 4 * HBUF, stream);

    pack_w<<<768, 256, 0, stream>>>(Whh0, WpHH0);
    pack_w<<<768, 256, 0, stream>>>(Wih1, WpIH1);
    pack_w<<<768, 256, 0, stream>>>(Whh1, WpHH1);

    gru_layer0<<<64, 256, 0, stream>>>(x, Wih0, bih0, bhh0, WpHH0, h0a, h0b, ys0, cnt0);

    gi1_gemm<<<(Tsz * Bsz * 16) / (16 * 8), 256, 0, stream>>>(ys0, WpIH1, bih1, gi1); // 23360 blocks

    gru_layer1<<<64, 256, 0, stream>>>(gi1, bhh1, WpHH1, h1a, h1b, cnt1);

    // T=365 (odd) -> final hidden state lives in the 'b' buffer
    classify<<<Bsz, 32, 0, stream>>>(h1b, Wout, bout, (float*)d_out);
}
